// AttentionLayer_24764781428881
// MI455X (gfx1250) — matrix operand order repaired and checked by reading, not compiled, not run on hardware
//
#include <hip/hip_runtime.h>
#include <hip/hip_bf16.h>

// Problem sizes (fixed by the reference)
#define BB 4
#define SS 2048
#define HH 16
#define DHD 64
#define DD 1024

typedef __attribute__((ext_vector_type(16))) __bf16 v16bf;
typedef __attribute__((ext_vector_type(8)))  float  v8f;
typedef __attribute__((ext_vector_type(4)))  unsigned int u32x4;
typedef __attribute__((ext_vector_type(8)))  int i32x8;
typedef __attribute__((ext_vector_type(4)))  int i32x4;

union FragB16 {
    v16bf v;
    uint4 q[2];
};

static __device__ __forceinline__ v8f wmma_bf16(v16bf a, v16bf b, v8f c) {
    // D = A(16x32 bf16) * B(32x16 bf16) + C(16x16 f32)
    return __builtin_amdgcn_wmma_f32_16x16x32_bf16(
        /*neg_a=*/false, a, /*neg_b=*/false, b,
        /*c_mod=*/(short)0, c, /*reuse_a=*/false, /*reuse_b=*/false);
}

// ---------------------------------------------------------------------------
// TDM: load a 128-row x 32-elem bf16 tile (row stride 1024 elems) from global
// into LDS, inserting 16B of padding after every 64B row so the LDS layout is
// [128][40] bf16 (80B stride, rows stay 16B-aligned).
// D# per CDNA5 ISA ch. 8 (group0: flags/lds/global addr; group1: dims/tile).
// Toolchain uses the 6-arg builtin: (g0, g1, g2, g3, g_extra, cpol).
// ---------------------------------------------------------------------------
static __device__ __forceinline__ void tdm_load_tile_128x32(
    const void* gsrc, unsigned lds_addr)
{
    unsigned long long ga = (unsigned long long)(uintptr_t)gsrc;
    u32x4 g0;
    g0.x = 1u;                                       // count=1 valid descriptor
    g0.y = lds_addr;                                 // lds_addr [63:32]
    g0.z = (unsigned)ga;                             // global_addr lo
    g0.w = (unsigned)((ga >> 32) & 0x1FFFFFFu)       // global_addr[56:32]
           | (2u << 30);                             // type=2 ("image")
    i32x8 g1;
    g1[0] = (int)((1u << 16)                         // data_size = 2 bytes
                | (1u << 20)                         // pad_enable
                | (3u << 22)                         // pad_interval: 16 DWORDs (64B)
                | (3u << 25));                       // pad_amount: 4 DWORDs (16B)
    g1[1] = (int)(1024u << 16);                      // tensor_dim0 = 1024 (lo16)
    g1[2] = 0;                                       // dim0 hi16 = 0, dim1 lo16 = 0
    g1[3] = (int)(0x10u | (32u << 16));              // tensor_dim1 = 1M rows; tile_dim0 = 32
    g1[4] = 128;                                     // tile_dim1 = 128; tile_dim2 = 0
    g1[5] = 1024;                                    // tensor_dim0_stride = 1024 elems
    g1[6] = 0;
    g1[7] = 0;
    i32x4 z4 = {0, 0, 0, 0};
    i32x8 z8 = {0, 0, 0, 0, 0, 0, 0, 0};
    __builtin_amdgcn_tensor_load_to_lds(g0, g1, z4, z4, z8, /*cpol=*/0);
}

// ---------------------------------------------------------------------------
// Pre-pass 1: x fp32 -> bf16 (row-major, unchanged layout)
// ---------------------------------------------------------------------------
__global__ __launch_bounds__(256) void cvt_x_kernel(
    const float* __restrict__ x, __bf16* __restrict__ xb)
{
    const size_t i = ((size_t)blockIdx.x * blockDim.x + threadIdx.x) * 4;
    float4 f = *(const float4*)(x + i);
    xb[i + 0] = (__bf16)f.x;
    xb[i + 1] = (__bf16)f.y;
    xb[i + 2] = (__bf16)f.z;
    xb[i + 3] = (__bf16)f.w;
}

// ---------------------------------------------------------------------------
// Pre-pass 2: W[k][n] fp32 -> Wt[w][n][k] bf16 (transposed), LDS-tiled 32x32.
// ---------------------------------------------------------------------------
__global__ __launch_bounds__(256) void transpose_w_kernel(
    const float* __restrict__ Wq, const float* __restrict__ Wk,
    const float* __restrict__ Wv, __bf16* __restrict__ Wt)
{
    const float* W = (blockIdx.z == 0) ? Wq : (blockIdx.z == 1) ? Wk : Wv;
    __bf16* dst = Wt + (size_t)blockIdx.z * DD * DD;
    __shared__ float tile[32][33];
    const int nb = blockIdx.x * 32;
    const int kb = blockIdx.y * 32;
    const int tx = threadIdx.x;         // 0..31
    const int ty = threadIdx.y;         // 0..7
#pragma unroll
    for (int j = 0; j < 4; ++j)
        tile[ty + 8 * j][tx] = W[(size_t)(kb + ty + 8 * j) * DD + nb + tx];
    __syncthreads();
#pragma unroll
    for (int j = 0; j < 4; ++j)
        dst[(size_t)(nb + ty + 8 * j) * DD + kb + tx] = (__bf16)tile[tx][ty + 8 * j];
}

// ---------------------------------------------------------------------------
// Kernel 1: fused QKV projection, TDM-fed + double-buffered LDS.
//   [8192 x 1024] x [1024 x 3072] (weight selected per 128-col tile)
// Block tile 128x128, K-chunk 32, 8 waves of 32x64 wave-tiles.
// Outputs: Q,K as bf16 [B,H,S,DH]; V as bf16 transposed [B,H,DH,S].
// ---------------------------------------------------------------------------
#define TM 128
#define TN 128
#define TK 32
#define LDSTR 40   // padded LDS row stride in bf16 elems (80 bytes)
#define NK (DD / TK)

__global__ __launch_bounds__(256) void qkv_proj_kernel(
    const __bf16* __restrict__ xb,   // [8192][1024] bf16
    const __bf16* __restrict__ Wt,   // [3][1024(n)][1024(k)] bf16 (transposed)
    const float* __restrict__ bq, const float* __restrict__ bk,
    const float* __restrict__ bv,
    __bf16* __restrict__ Qb, __bf16* __restrict__ Kb, __bf16* __restrict__ Vt)
{
    __shared__ __attribute__((aligned(16))) __bf16 As[2][TM][LDSTR];
    __shared__ __attribute__((aligned(16))) __bf16 Bt[2][TN][LDSTR];

    const int tid  = threadIdx.x;
    const int lane = tid & 31;
    const int wv   = tid >> 5;       // wave 0..7
    const int half = lane >> 4;      // 0/1
    const int lr   = lane & 15;
    const int mw   = wv >> 1;        // 0..3  -> M offset mw*32
    const int nw   = wv & 1;         // 0..1  -> N offset nw*64

    const int m0    = blockIdx.x * TM;
    const int ntile = blockIdx.y;            // 0..23 over 3*1024 cols
    const int wid   = ntile >> 3;            // 0=Q,1=K,2=V
    const int n0    = (ntile & 7) * TN;      // column within this weight

    const float* bias = (wid == 0) ? bq : (wid == 1) ? bk : bv;

    const __bf16* Ag = xb + (size_t)m0 * DD;                       // A tile base (k0 added)
    const __bf16* Bg = Wt + (size_t)wid * DD * DD + (size_t)n0 * DD;

    const unsigned asBase = (unsigned)(uintptr_t)&As[0][0][0];
    const unsigned btBase = (unsigned)(uintptr_t)&Bt[0][0][0];
    const unsigned bufBytes = TM * LDSTR * 2;   // 10240

    v8f acc[2][4] = {};

    // prefetch tile 0 (wave 0 drives the TDM; TENSORcnt is per-wave)
    if (wv == 0) {
        tdm_load_tile_128x32(Ag, asBase);
        tdm_load_tile_128x32(Bg, btBase);
    }

    for (int it = 0; it < NK; ++it) {
        if (wv == 0) __builtin_amdgcn_s_wait_tensorcnt(0);
        __syncthreads();    // tile `it` visible to all waves; prior reads done

        if (wv == 0 && (it + 1) < NK) {
            const int k0n = (it + 1) * TK;
            const unsigned boff = (unsigned)((it + 1) & 1) * bufBytes;
            tdm_load_tile_128x32(Ag + k0n, asBase + boff);
            tdm_load_tile_128x32(Bg + k0n, btBase + boff);
        }

        const __bf16 (*Ab)[LDSTR] = As[it & 1];
        const __bf16 (*Bb)[LDSTR] = Bt[it & 1];

        // ---- A fragments (16x32): lane<16 holds K 0-7/16-23, lane>=16: 8-15/24-31 ----
        FragB16 afr[2];
#pragma unroll
        for (int mf = 0; mf < 2; ++mf) {
            const int row = mw * 32 + mf * 16 + lr;
            afr[mf].q[0] = *(const uint4*)&Ab[row][half * 8];
            afr[mf].q[1] = *(const uint4*)&Ab[row][16 + half * 8];
        }
        // ---- B fragments (32x16): lane = column n, lane<16 holds K 0-7/16-23, lane>=16: 8-15/24-31 ----
#pragma unroll
        for (int nf = 0; nf < 4; ++nf) {
            FragB16 bfr;
            const int n = nw * 64 + nf * 16 + lr;
            bfr.q[0] = *(const uint4*)&Bb[n][half * 8];
            bfr.q[1] = *(const uint4*)&Bb[n][16 + half * 8];
#pragma unroll
            for (int mf = 0; mf < 2; ++mf)
                acc[mf][nf] = wmma_bf16(afr[mf].v, bfr.v, acc[mf][nf]);
        }
    }

    // ---- epilogue: bias add, convert to bf16, scatter into head layouts ----
#pragma unroll
    for (int nf = 0; nf < 4; ++nf) {
        const int jw = n0 + nw * 64 + nf * 16 + lr;   // col within this weight
        const float bj = bias[jw];
        const int h  = jw >> 6;
        const int dh = jw & 63;
#pragma unroll
        for (int mf = 0; mf < 2; ++mf) {
#pragma unroll
            for (int i = 0; i < 8; ++i) {
                const int r = m0 + mw * 32 + mf * 16 + i + 8 * half;  // C layout: M = i + 8*half
                const int b = r >> 11;          // /2048
                const int s = r & 2047;
                const float val = acc[mf][nf][i] + bj;
                const size_t bh = (size_t)(b * HH + h);
                if (wid == 2) {
                    Vt[(bh * DHD + dh) * SS + s] = (__bf16)val;        // V transposed [B,H,DH,S]
                } else {
                    __bf16* dst = (wid == 0) ? Qb : Kb;                // [B,H,S,DH]
                    dst[(bh * SS + s) * DHD + dh] = (__bf16)val;
                }
            }
        }
    }
}

// ---------------------------------------------------------------------------
// Kernel 2: flash attention. 1 block = (b,h) x 64 queries, 4 waves x 16 q.
// Online softmax over 32-key chunks; P re-staged through per-wave LDS slab
// to convert WMMA C-layout -> A-layout for the P*V GEMM.
// ---------------------------------------------------------------------------
#define PSTR 40

__global__ __launch_bounds__(128) void attn_kernel(
    const __bf16* __restrict__ Qb, const __bf16* __restrict__ Kb,
    const __bf16* __restrict__ Vt, float* __restrict__ out)
{
    __shared__ __attribute__((aligned(16))) __bf16 Ps[4][16][PSTR];

    const int tid  = threadIdx.x;
    const int lane = tid & 31;
    const int wv   = tid >> 5;        // 0..3
    const int half = lane >> 4;
    const int lr   = lane & 15;

    const int bh = blockIdx.x >> 5;   // b*16 + h
    const int qt = blockIdx.x & 31;
    const int b  = bh >> 4;
    const int h  = bh & 15;
    const int q0 = qt * 64 + wv * 16;

    const __bf16* Qh = Qb + ((size_t)bh * SS + q0) * DHD;
    const __bf16* Kh = Kb + (size_t)bh * SS * DHD;
    const __bf16* Vh = Vt + (size_t)bh * DHD * SS;

    // Q fragments for this wave's 16 queries (DH = 64 -> 2 K-chunks of 32)
    FragB16 qf[2];
#pragma unroll
    for (int f = 0; f < 2; ++f) {
        const __bf16* p = Qh + (size_t)lr * DHD + f * 32 + half * 8;
        qf[f].q[0] = *(const uint4*)p;
        qf[f].q[1] = *(const uint4*)(p + 16);
    }

    float mrun[8], lrun[8];
#pragma unroll
    for (int i = 0; i < 8; ++i) { mrun[i] = -3.0e38f; lrun[i] = 0.0f; }
    v8f cacc[4] = {};   // 16 queries x 64 dh

    for (int kk0 = 0; kk0 < SS; kk0 += 32) {
        // ---- scores: 16q x 32k, reduce over DH=64 ----
        v8f sc[2] = {};
#pragma unroll
        for (int nf = 0; nf < 2; ++nf) {
#pragma unroll
            for (int ks = 0; ks < 2; ++ks) {
                FragB16 kf;   // B-frag: lane = key, halves pick d 0-7/16-23 / 8-15/24-31 of this chunk
                const __bf16* p = Kh + (size_t)(kk0 + nf * 16 + lr) * DHD + ks * 32 + half * 8;
                kf.q[0] = *(const uint4*)p;
                kf.q[1] = *(const uint4*)(p + 16);
                sc[nf] = wmma_bf16(qf[ks].v, kf.v, sc[nf]);
            }
        }

        // ---- online softmax (rows live across a 16-lane group; xor-reduce) ----
        const float scale = 0.125f;   // 1/sqrt(64)
#pragma unroll
        for (int i = 0; i < 8; ++i) {
            float s0 = sc[0][i] * scale;
            float s1 = sc[1][i] * scale;
            float mx = fmaxf(s0, s1);
#pragma unroll
            for (int off = 8; off >= 1; off >>= 1)
                mx = fmaxf(mx, __shfl_xor(mx, off, 32));
            const float mnew  = fmaxf(mrun[i], mx);
            const float alpha = __expf(mrun[i] - mnew);
            mrun[i] = mnew;
            const float p0 = __expf(s0 - mnew);
            const float p1 = __expf(s1 - mnew);
            float rs = p0 + p1;
#pragma unroll
            for (int off = 8; off >= 1; off >>= 1)
                rs += __shfl_xor(rs, off, 32);
            lrun[i] = lrun[i] * alpha + rs;
            sc[0][i] = p0;
            sc[1][i] = p1;
#pragma unroll
            for (int nf = 0; nf < 4; ++nf) cacc[nf][i] *= alpha;
        }

        // ---- P: C-layout -> LDS -> A-layout ----
#pragma unroll
        for (int nf = 0; nf < 2; ++nf)
#pragma unroll
            for (int i = 0; i < 8; ++i)
                Ps[wv][i + 8 * half][nf * 16 + lr] = (__bf16)sc[nf][i];
        asm volatile("s_wait_dscnt 0" ::: "memory");

        FragB16 pa;
        pa.q[0] = *(const uint4*)&Ps[wv][lr][half * 8];
        pa.q[1] = *(const uint4*)&Ps[wv][lr][16 + half * 8];

        // ---- ctx += P(16x32) * V(32x64); V stored transposed [dh][s] ----
#pragma unroll
        for (int nf = 0; nf < 4; ++nf) {
            FragB16 vf;
            const __bf16* p = Vh + (size_t)(nf * 16 + lr) * SS + kk0 + half * 8;
            vf.q[0] = *(const uint4*)p;
            vf.q[1] = *(const uint4*)(p + 16);
            cacc[nf] = wmma_bf16(pa.v, vf.v, cacc[nf]);
        }
    }

    // ---- normalize and write fp32 output [B,S,D] ----
#pragma unroll
    for (int nf = 0; nf < 4; ++nf) {
        const int j = h * 64 + nf * 16 + lr;
#pragma unroll
        for (int i = 0; i < 8; ++i) {
            const int s = q0 + i + 8 * half;
            out[((size_t)b * SS + s) * DD + j] = cacc[nf][i] / lrun[i];
        }
    }
}

// ---------------------------------------------------------------------------
extern "C" void kernel_launch(void* const* d_in, const int* in_sizes, int n_in,
                              void* d_out, int out_size, void* d_ws, size_t ws_size,
                              hipStream_t stream) {
    const float* x  = (const float*)d_in[0];
    const float* Wq = (const float*)d_in[1];
    const float* bq = (const float*)d_in[2];
    const float* Wk = (const float*)d_in[3];
    const float* bk = (const float*)d_in[4];
    const float* Wv = (const float*)d_in[5];
    const float* bv = (const float*)d_in[6];
    float* out = (float*)d_out;

    // Workspace layout (bf16):
    //   xb  [B*S, D]            8,388,608 elems
    //   Wt  [3, D(n), D(k)]     3,145,728 elems
    //   Qb,Kb [B,H,S,DH]        8,388,608 elems each
    //   Vt  [B,H,DH,S]          8,388,608 elems
    const size_t xElems = (size_t)BB * SS * DD;
    const size_t wElems = (size_t)3 * DD * DD;
    const size_t hElems = (size_t)BB * HH * SS * DHD;
    __bf16* xb = (__bf16*)d_ws;
    __bf16* Wt = xb + xElems;
    __bf16* Qb = Wt + wElems;
    __bf16* Kb = Qb + hElems;
    __bf16* Vt = Kb + hElems;

    cvt_x_kernel<<<dim3((unsigned)(xElems / 4 / 256)), 256, 0, stream>>>(x, xb);
    transpose_w_kernel<<<dim3(DD / 32, DD / 32, 3), dim3(32, 8), 0, stream>>>(Wq, Wk, Wv, Wt);

    dim3 gproj((BB * SS) / TM, (3 * DD) / TN);         // 64 x 24
    qkv_proj_kernel<<<gproj, 256, 0, stream>>>(xb, Wt, bq, bk, bv, Qb, Kb, Vt);

    dim3 gattn(BB * HH * (SS / 64));                   // 2048 blocks
    attn_kernel<<<gattn, 128, 0, stream>>>(Qb, Kb, Vt, out);
}
